// QP_MSA_25709674234118
// MI455X (gfx1250) — matrix operand order repaired and checked by reading, not compiled, not run on hardware
//
#include <hip/hip_runtime.h>
#include <math.h>

// ---------------------------------------------------------------------------
// IGAB channel attention for MI455X (gfx1250, wave32, WMMA bf16 16x16x32)
// ---------------------------------------------------------------------------
#define HEADS 8
#define DHEAD 40
#define CDIM  320
#define HH    256
#define WWID  256
#define BATCH 2
#define NTOK  65536            // tokens per batch image
#define ROWS  131072           // total tokens (BATCH * NTOK)

typedef __bf16 bf16_t;
typedef bf16_t bf16x16 __attribute__((ext_vector_type(16)));
typedef float  f32x8  __attribute__((ext_vector_type(8)));

union Frag { unsigned u[8]; bf16x16 v; };

__device__ __forceinline__ unsigned short f2bf(float f) {
  union { float f; unsigned u; } x; x.f = f;
  unsigned r = x.u + 0x7fffu + ((x.u >> 16) & 1u);
  return (unsigned short)(r >> 16);
}
__device__ __forceinline__ float bf2f(unsigned short h) {
  union { unsigned u; float f; } x; x.u = ((unsigned)h) << 16; return x.f;
}

// ---------------------------------------------------------------------------
// Pack a 320x320 fp32 row-major matrix into WMMA B-operand bf16 tiles.
// Tile (nt,kt) -> 32 lanes x 16 bf16 contiguous.  B element i of lane L:
//   col = nt*16 + (L&15),  K = kt*32 + (L>=16 ? 8 : 0) + i + (i>=8 ? 8 : 0)
// grid.x = 200 (20 n-tiles x 10 k-tiles), block = 32
// ---------------------------------------------------------------------------
__global__ void pack_b_kernel(const float* __restrict__ src,
                              unsigned short* __restrict__ dst) {
  int t = blockIdx.x;                    // nt*10 + kt
  int nt = t / 10, kt = t % 10;
  int lane = threadIdx.x;
  int n  = nt * 16 + (lane & 15);
  int kb = kt * 32 + ((lane >= 16) ? 8 : 0);
  unsigned short* o = dst + (size_t)t * 512 + (size_t)lane * 16;
#pragma unroll
  for (int i = 0; i < 16; ++i) o[i] = f2bf(src[(kb + i + (i >= 8 ? 8 : 0)) * CDIM + n]);
}

// ---------------------------------------------------------------------------
// Fused QKV projection: q = x@Wq, k = x@Wk, v = x@Wv; also vi = v*illu.
// Block = 128 thr (4 waves), 64-row tile.  x tile staged in LDS as bf16.
// Two N-tiles per iteration (shared A frags, two WMMA chains) with
// double-buffered B-fragment loads so loads overlap matrix math.
// ---------------------------------------------------------------------------
__global__ void __launch_bounds__(128) gemm_qkv_kernel(
    const float* __restrict__ x, const float* __restrict__ illu,
    const unsigned short* __restrict__ wpack,
    unsigned short* __restrict__ qws, unsigned short* __restrict__ kws,
    unsigned short* __restrict__ viws, unsigned short* __restrict__ vws) {
  __shared__ unsigned short xl[64 * CDIM];          // 40 KB
  const int tid = threadIdx.x;
  const long row0 = (long)blockIdx.x * 64;

  // prefetch the illu rows consumed by the epilogue (global_prefetch_b8)
  __builtin_prefetch(illu + row0 * CDIM + (size_t)tid * 160, 0, 1);

  // stage x tile fp32 -> bf16 (vectorized float4 loads, uint2 LDS stores)
  const float4* xs = (const float4*)(x + row0 * CDIM);
#pragma unroll 4
  for (int j = 0; j < 40; ++j) {
    int i4 = tid + j * 128;                          // 5120 float4
    float4 v4 = xs[i4];
    uint2 p;
    p.x = (unsigned)f2bf(v4.x) | ((unsigned)f2bf(v4.y) << 16);
    p.y = (unsigned)f2bf(v4.z) | ((unsigned)f2bf(v4.w) << 16);
    ((uint2*)xl)[i4] = p;
  }
  __syncthreads();

  const int wv = tid >> 5, lane = tid & 31;
  const int l15 = lane & 15;
  const int hi8 = (lane >= 16) ? 8 : 0;
  const int mrow = (wv << 4) + l15;

  // preload all 10 A fragments for this wave's 16-row strip
  Frag a[10];
#pragma unroll
  for (int kt = 0; kt < 10; ++kt) {
    const unsigned short* bse = &xl[mrow * CDIM + kt * 32 + hi8];
#pragma unroll
    for (int p = 0; p < 8; ++p) {
      int koff = (p < 4) ? 2 * p : 2 * p + 8;
      a[kt].u[p] = *(const unsigned*)(bse + koff);
    }
  }

  for (int mat = 0; mat < 3; ++mat) {
    for (int ntp = 0; ntp < 10; ++ntp) {
      const int nt0 = ntp * 2;
      f32x8 acc0 = {0.f, 0.f, 0.f, 0.f, 0.f, 0.f, 0.f, 0.f};
      f32x8 acc1 = {0.f, 0.f, 0.f, 0.f, 0.f, 0.f, 0.f, 0.f};
      const unsigned short* bp0 =
          wpack + ((size_t)(mat * 200 + nt0 * 10) * 512) + (size_t)lane * 16;
      const unsigned short* bp1 = bp0 + (size_t)10 * 512;
      Frag c0, c1;
      c0.v = *(const bf16x16*)(bp0);
      c1.v = *(const bf16x16*)(bp1);
#pragma unroll
      for (int kt = 0; kt < 10; ++kt) {
        Frag n0, n1;
        if (kt < 9) {                    // double-buffer: fetch next K-step
          n0.v = *(const bf16x16*)(bp0 + (size_t)(kt + 1) * 512);
          n1.v = *(const bf16x16*)(bp1 + (size_t)(kt + 1) * 512);
        }
        acc0 = __builtin_amdgcn_wmma_f32_16x16x32_bf16(
            false, a[kt].v, false, c0.v, (short)0, acc0, false, false);
        acc1 = __builtin_amdgcn_wmma_f32_16x16x32_bf16(
            false, a[kt].v, false, c1.v, (short)0, acc1, false, false);
        if (kt < 9) { c0 = n0; c1 = n1; }
      }
#pragma unroll
      for (int t2 = 0; t2 < 2; ++t2) {
        const f32x8 acc = t2 ? acc1 : acc0;
        const int col = (nt0 + t2) * 16 + l15;
#pragma unroll
        for (int j = 0; j < 8; ++j) {
          long r = row0 + (wv << 4) + j + hi8;
          long off = r * CDIM + col;
          float v = acc[j];
          if (mat == 0)      qws[off] = f2bf(v);
          else if (mat == 1) kws[off] = f2bf(v);
          else {
            float il = illu[off];
            viws[off] = f2bf(v * il);
            vws[off]  = f2bf(v);
          }
        }
      }
    }
  }
}

// ---------------------------------------------------------------------------
// Per-column sum of squares of q/k (for the F.normalize norms).
// grid = (2048, 2), block = 320; fp32 atomics into sumsq[b*320+c].
// ---------------------------------------------------------------------------
__global__ void colsumsq_kernel(const unsigned short* __restrict__ q,
                                const unsigned short* __restrict__ k,
                                float* __restrict__ sqq,
                                float* __restrict__ sqk) {
  const int c = threadIdx.x;
  const long r0 = (long)blockIdx.x * 64;
  const unsigned short* src = blockIdx.y ? k : q;
  float* dst = blockIdx.y ? sqk : sqq;
  float acc = 0.f;
  for (int r = 0; r < 64; ++r) {
    float v = bf2f(src[(r0 + r) * CDIM + c]);
    acc += v * v;
  }
  atomicAdd(&dst[(int)(r0 / NTOK) * CDIM + c], acc);
}

// ---------------------------------------------------------------------------
// Split-K Gram: G[b,h,d,e] += sum over a 256-token chunk of k[.,d]*q[.,e].
// 48x48 padded tiles (3x3 WMMA tiles), A = K^T, B = Q, gathered from
// channel-major LDS staging.  A frags hoisted per td; B gathers pipelined.
// grid = (256, 8, 2), block = 64 (2 waves).
// ---------------------------------------------------------------------------
#define GTOK 256

__device__ __forceinline__ void gather_a(Frag& f, const unsigned short* base) {
#pragma unroll
  for (int p = 0; p < 8; ++p) {
    int koff = (p < 4) ? 2 * p : 2 * p + 8;
    f.u[p] = *(const unsigned*)(base + koff);
  }
}
__device__ __forceinline__ void gather_b(Frag& f, const unsigned short* base) {
#pragma unroll
  for (int p = 0; p < 8; ++p) f.u[p] = *(const unsigned*)(base + ((p < 4) ? 2 * p : 2 * p + 8));
}

__global__ void __launch_bounds__(64) gram_kernel(
    const unsigned short* __restrict__ qws,
    const unsigned short* __restrict__ kws,
    float* __restrict__ G) {
  __shared__ unsigned short kT[48 * GTOK];          // 24 KB
  __shared__ unsigned short qT[48 * GTOK];          // 24 KB
  const int tid = threadIdx.x;
  const int h = blockIdx.y, b = blockIdx.z;
  const long n0 = (long)b * NTOK + (long)blockIdx.x * GTOK;
  const int cbase = h * DHEAD;

  for (int i = tid; i < 8 * GTOK; i += 64) {        // zero pad channels 40..47
    kT[40 * GTOK + i] = 0;
    qT[40 * GTOK + i] = 0;
  }
  for (int i = tid; i < GTOK * DHEAD; i += 64) {    // stage channel-major
    int t = i / DHEAD, c = i % DHEAD;
    long g = (n0 + t) * CDIM + cbase + c;
    kT[c * GTOK + t] = kws[g];
    qT[c * GTOK + t] = qws[g];
  }
  __syncthreads();

  const int wv = tid >> 5, lane = tid & 31;
  const int l15 = lane & 15;
  const int hi8 = (lane >= 16) ? 8 : 0;
  const int hi16 = (lane >= 16) ? 16 : 0;
  const int tb = wv * 128;                          // 128 tokens per wave

  for (int td = 0; td < 3; ++td) {
    // A fragments for this row tile (reused across all 3 column tiles)
    Frag afr[4];
    const unsigned short* ab = &kT[(td * 16 + l15) * GTOK + tb + hi8];
#pragma unroll
    for (int ks = 0; ks < 4; ++ks) gather_a(afr[ks], ab + ks * 32);

    for (int te = 0; te < 3; ++te) {
      f32x8 acc = {0.f, 0.f, 0.f, 0.f, 0.f, 0.f, 0.f, 0.f};
      const unsigned short* bb = &qT[(te * 16 + l15) * GTOK + tb + hi8];
      Frag b0;
      gather_b(b0, bb);
#pragma unroll
      for (int ks = 0; ks < 4; ++ks) {
        Frag bn;
        if (ks < 3) gather_b(bn, bb + (ks + 1) * 32);
        acc = __builtin_amdgcn_wmma_f32_16x16x32_bf16(
            false, afr[ks].v, false, b0.v, (short)0, acc, false, false);
        if (ks < 3) b0 = bn;
      }
      const int e = te * 16 + l15;
#pragma unroll
      for (int j = 0; j < 8; ++j) {
        int d = td * 16 + j + hi8;
        if (d < DHEAD && e < DHEAD)
          atomicAdd(&G[((b * HEADS + h) * DHEAD + d) * DHEAD + e], acc[j]);
      }
    }
  }
}

// ---------------------------------------------------------------------------
// attn = softmax(rescale * G / (||k_d|| ||q_e||)) over e; then fold into Wp:
//   M[b][h*40+e][c'] = sum_d attn[h,d,e] * Wp[h*40+d][c']
// grid = (8, 2), block = 320
// ---------------------------------------------------------------------------
__global__ void softmax_m_kernel(const float* __restrict__ G,
                                 const float* __restrict__ sqq,
                                 const float* __restrict__ sqk,
                                 const float* __restrict__ rescale,
                                 const float* __restrict__ Wp,
                                 float* __restrict__ M) {
  const int h = blockIdx.x, b = blockIdx.y;
  const int tid = threadIdx.x;
  __shared__ float nq[DHEAD], nk[DHEAD];
  __shared__ float attn[DHEAD][DHEAD];
  if (tid < DHEAD) {
    nq[tid] = fmaxf(sqrtf(fmaxf(sqq[b * CDIM + h * DHEAD + tid], 0.f)), 1e-12f);
    nk[tid] = fmaxf(sqrtf(fmaxf(sqk[b * CDIM + h * DHEAD + tid], 0.f)), 1e-12f);
  }
  __syncthreads();
  const float resc = rescale[h];
  if (tid < DHEAD) {
    const int d = tid;
    const float* g = &G[((b * HEADS + h) * DHEAD + d) * DHEAD];
    float row[DHEAD];
    float mx = -1e30f;
    for (int e = 0; e < DHEAD; ++e) {
      row[e] = g[e] / (nk[d] * nq[e]) * resc;
      mx = fmaxf(mx, row[e]);
    }
    float s = 0.f;
    for (int e = 0; e < DHEAD; ++e) { row[e] = __expf(row[e] - mx); s += row[e]; }
    float inv = 1.f / s;
    for (int e = 0; e < DHEAD; ++e) attn[d][e] = row[e] * inv;
  }
  __syncthreads();
  for (int idx = tid; idx < DHEAD * CDIM; idx += blockDim.x) {
    int e = idx / CDIM, cp = idx % CDIM;
    float s = 0.f;
    for (int d = 0; d < DHEAD; ++d)
      s += attn[d][e] * Wp[(h * DHEAD + d) * CDIM + cp];
    M[(size_t)b * CDIM * CDIM + (size_t)(h * DHEAD + e) * CDIM + cp] = s;
  }
}

// ---------------------------------------------------------------------------
// Depthwise 3x3 (SAME) + exact GELU on v_inp -> t1 (bf16)
// ---------------------------------------------------------------------------
__global__ void dwconv_gelu_kernel(const unsigned short* __restrict__ v,
                                   const float* __restrict__ w,
                                   unsigned short* __restrict__ t1) {
  long id = (long)blockIdx.x * blockDim.x + threadIdx.x;
  if (id >= (long)BATCH * HH * WWID * CDIM) return;
  int c = (int)(id % CDIM);
  long r = id / CDIM;
  int x = (int)(r % WWID);
  long r2 = r / WWID;
  int y = (int)(r2 % HH);
  int b = (int)(r2 / HH);
  float s = 0.f;
#pragma unroll
  for (int dy = -1; dy <= 1; ++dy)
#pragma unroll
    for (int dx = -1; dx <= 1; ++dx) {
      int yy = y + dy, xx = x + dx;
      if (yy >= 0 && yy < HH && xx >= 0 && xx < WWID)
        s += bf2f(v[(((long)b * HH + yy) * WWID + xx) * CDIM + c]) *
             w[c * 9 + (dy + 1) * 3 + (dx + 1)];
    }
  float g = 0.5f * s * (1.f + erff(s * 0.70710678118654752f));
  t1[id] = f2bf(g);
}

// Depthwise 3x3 on t1 -> writes out_p (fp32) into d_out (full overwrite)
__global__ void dwconv_out_kernel(const unsigned short* __restrict__ t1,
                                  const float* __restrict__ w,
                                  float* __restrict__ out) {
  long id = (long)blockIdx.x * blockDim.x + threadIdx.x;
  if (id >= (long)BATCH * HH * WWID * CDIM) return;
  int c = (int)(id % CDIM);
  long r = id / CDIM;
  int x = (int)(r % WWID);
  long r2 = r / WWID;
  int y = (int)(r2 % HH);
  int b = (int)(r2 / HH);
  float s = 0.f;
#pragma unroll
  for (int dy = -1; dy <= 1; ++dy)
#pragma unroll
    for (int dx = -1; dx <= 1; ++dx) {
      int yy = y + dy, xx = x + dx;
      if (yy >= 0 && yy < HH && xx >= 0 && xx < WWID)
        s += bf2f(t1[(((long)b * HH + yy) * WWID + xx) * CDIM + c]) *
             w[c * 9 + (dy + 1) * 3 + (dx + 1)];
    }
  out[id] = s;
}

// ---------------------------------------------------------------------------
// Final GEMM: d_out += vi @ M[batch] + bp   (vi bf16 staged in LDS)
// Same 2-wide / double-buffered WMMA structure as the QKV GEMM.
// ---------------------------------------------------------------------------
__global__ void __launch_bounds__(128) gemm_out_kernel(
    const unsigned short* __restrict__ viws,
    const unsigned short* __restrict__ mpack,
    const float* __restrict__ bias, float* __restrict__ out) {
  __shared__ unsigned short xl[64 * CDIM];
  const int tid = threadIdx.x;
  const long row0 = (long)blockIdx.x * 64;
  const int bb = (int)(row0 / NTOK);

  const uint4* vs = (const uint4*)(viws + row0 * CDIM);
#pragma unroll 4
  for (int j = 0; j < 20; ++j) {
    int i8 = tid + j * 128;
    ((uint4*)xl)[i8] = vs[i8];
  }
  __syncthreads();

  const int wv = tid >> 5, lane = tid & 31;
  const int l15 = lane & 15;
  const int hi8 = (lane >= 16) ? 8 : 0;
  const int mrow = (wv << 4) + l15;

  Frag a[10];
#pragma unroll
  for (int kt = 0; kt < 10; ++kt) {
    const unsigned short* bse = &xl[mrow * CDIM + kt * 32 + hi8];
#pragma unroll
    for (int p = 0; p < 8; ++p) {
      int koff = (p < 4) ? 2 * p : 2 * p + 8;
      a[kt].u[p] = *(const unsigned*)(bse + koff);
    }
  }

  for (int ntp = 0; ntp < 10; ++ntp) {
    const int nt0 = ntp * 2;
    f32x8 acc0 = {0.f, 0.f, 0.f, 0.f, 0.f, 0.f, 0.f, 0.f};
    f32x8 acc1 = {0.f, 0.f, 0.f, 0.f, 0.f, 0.f, 0.f, 0.f};
    const unsigned short* bp0 =
        mpack + ((size_t)(bb * 200 + nt0 * 10) * 512) + (size_t)lane * 16;
    const unsigned short* bp1 = bp0 + (size_t)10 * 512;
    Frag c0, c1;
    c0.v = *(const bf16x16*)(bp0);
    c1.v = *(const bf16x16*)(bp1);
#pragma unroll
    for (int kt = 0; kt < 10; ++kt) {
      Frag n0, n1;
      if (kt < 9) {
        n0.v = *(const bf16x16*)(bp0 + (size_t)(kt + 1) * 512);
        n1.v = *(const bf16x16*)(bp1 + (size_t)(kt + 1) * 512);
      }
      acc0 = __builtin_amdgcn_wmma_f32_16x16x32_bf16(
          false, a[kt].v, false, c0.v, (short)0, acc0, false, false);
      acc1 = __builtin_amdgcn_wmma_f32_16x16x32_bf16(
          false, a[kt].v, false, c1.v, (short)0, acc1, false, false);
      if (kt < 9) { c0 = n0; c1 = n1; }
    }
#pragma unroll
    for (int t2 = 0; t2 < 2; ++t2) {
      const f32x8 acc = t2 ? acc1 : acc0;
      const int col = (nt0 + t2) * 16 + l15;
#pragma unroll
      for (int j = 0; j < 8; ++j) {
        long r = row0 + (wv << 4) + j + hi8;
        long off = r * CDIM + col;
        out[off] = out[off] + acc[j] + bias[col];
      }
    }
  }
}

// ---------------------------------------------------------------------------
// Host launcher.  Workspace layout (bytes); needs ~421 MB of d_ws.
// ---------------------------------------------------------------------------
extern "C" void kernel_launch(void* const* d_in, const int* in_sizes, int n_in,
                              void* d_out, int out_size, void* d_ws, size_t ws_size,
                              hipStream_t stream) {
  (void)in_sizes; (void)n_in; (void)out_size; (void)ws_size;
  const float* x_in   = (const float*)d_in[0];
  const float* illu   = (const float*)d_in[1];
  const float* Wq     = (const float*)d_in[2];
  const float* Wk     = (const float*)d_in[3];
  const float* Wv     = (const float*)d_in[4];
  const float* rescale= (const float*)d_in[5];
  const float* Wp     = (const float*)d_in[6];
  const float* bp     = (const float*)d_in[7];
  const float* pw1    = (const float*)d_in[8];
  const float* pw2    = (const float*)d_in[9];
  float* out = (float*)d_out;

  char* ws = (char*)d_ws;
  const size_t SZB = (size_t)ROWS * CDIM * 2;            // 83,886,080 bytes
  unsigned short* q_ws  = (unsigned short*)(ws + 0 * SZB);
  unsigned short* k_ws  = (unsigned short*)(ws + 1 * SZB);
  unsigned short* vi_ws = (unsigned short*)(ws + 2 * SZB);
  unsigned short* v_ws  = (unsigned short*)(ws + 3 * SZB);
  unsigned short* t1_ws = (unsigned short*)(ws + 4 * SZB);
  size_t off = 5 * SZB;
  unsigned short* wpack = (unsigned short*)(ws + off); off += (size_t)3 * 200 * 512 * 2;
  unsigned short* mpack = (unsigned short*)(ws + off); off += (size_t)2 * 200 * 512 * 2;
  float* G_ws   = (float*)(ws + off); off += (size_t)BATCH * HEADS * DHEAD * DHEAD * 4;
  float* M_ws   = (float*)(ws + off); off += (size_t)BATCH * CDIM * CDIM * 4;
  float* sqq_ws = (float*)(ws + off); off += (size_t)BATCH * CDIM * 4;
  float* sqk_ws = (float*)(ws + off); off += (size_t)BATCH * CDIM * 4;

  // zero the accumulators (graph-capturable async memsets)
  hipMemsetAsync(G_ws,   0, (size_t)BATCH * HEADS * DHEAD * DHEAD * 4, stream);
  hipMemsetAsync(sqq_ws, 0, (size_t)BATCH * CDIM * 4, stream);
  hipMemsetAsync(sqk_ws, 0, (size_t)BATCH * CDIM * 4, stream);

  // 1) pack Wq/Wk/Wv into WMMA B-operand layout
  pack_b_kernel<<<200, 32, 0, stream>>>(Wq, wpack + (size_t)0 * 200 * 512);
  pack_b_kernel<<<200, 32, 0, stream>>>(Wk, wpack + (size_t)1 * 200 * 512);
  pack_b_kernel<<<200, 32, 0, stream>>>(Wv, wpack + (size_t)2 * 200 * 512);

  // 2) fused QKV projection (WMMA bf16)
  gemm_qkv_kernel<<<ROWS / 64, 128, 0, stream>>>(x_in, illu, wpack,
                                                 q_ws, k_ws, vi_ws, v_ws);

  // 3) column sum-of-squares for the L2 norms
  colsumsq_kernel<<<dim3(ROWS / 64, 2), 320, 0, stream>>>(q_ws, k_ws,
                                                          sqq_ws, sqk_ws);

  // 4) split-K Gram matrices G = k^T q per (batch, head) (WMMA + atomics)
  gram_kernel<<<dim3(NTOK / GTOK, HEADS, BATCH), 64, 0, stream>>>(q_ws, k_ws, G_ws);

  // 5) softmax + fold attention into Wp -> M[b] (320x320)
  softmax_m_kernel<<<dim3(HEADS, BATCH), 320, 0, stream>>>(
      G_ws, sqq_ws, sqk_ws, rescale, Wp, M_ws);

  // 6) pack M into WMMA B layout (per batch)
  pack_b_kernel<<<200, 32, 0, stream>>>(M_ws, mpack);
  pack_b_kernel<<<200, 32, 0, stream>>>(M_ws + (size_t)CDIM * CDIM,
                                        mpack + (size_t)200 * 512);

  // 7) positional branch: dwconv -> gelu -> dwconv, overwrites d_out
  const long nel = (long)BATCH * HH * WWID * CDIM;
  const int dwblocks = (int)((nel + 255) / 256);
  dwconv_gelu_kernel<<<dwblocks, 256, 0, stream>>>(v_ws, pw1, t1_ws);
  dwconv_out_kernel<<<dwblocks, 256, 0, stream>>>(t1_ws, pw2, out);

  // 8) final projection: d_out += vi @ M + bp (WMMA bf16)
  gemm_out_kernel<<<ROWS / 64, 128, 0, stream>>>(vi_ws, mpack, bp, out);
}
